// GCGRUCell_epsilon_19473381720845
// MI455X (gfx1250) — matrix operand order repaired and checked by reading, not compiled, not run on hardware
//
#include <hip/hip_runtime.h>
#include <hip/hip_bf16.h>

// ---------------------------------------------------------------------------
// GCGRU cell (diffusion-GRU + transformer epsilon) for gfx1250 / MI455X.
// All GEMMs via v_wmma_f32_16x16x32_bf16 (bf16 inputs, f32 accumulate).
// Register-cached operands: A tiles live in VGPRs and are reused across
// output tiles to cut global traffic (gate: 8 WMMA per z load).
// ---------------------------------------------------------------------------

#define Bb   256   // batch
#define DIN  64
#define NUc  256   // hidden / d
#define Nn   207   // nodes
#define KCc  1280  // (ORDER+1)*(DIN+NU)
#define NP   224   // padded node pitch (multiple of 32)
#define NBr  (Nn * Bb)  // 52992 rows of the (N,B,d) tensor

typedef __attribute__((ext_vector_type(16))) __bf16        v16bf;
typedef __attribute__((ext_vector_type(8)))  float         v8f;
typedef __attribute__((ext_vector_type(8)))  unsigned int  v8u;

// ----- helpers -------------------------------------------------------------

__device__ __forceinline__ unsigned short f2bf(float f) {
  unsigned int u = __builtin_bit_cast(unsigned int, f);
  unsigned int r = u + 0x7FFFu + ((u >> 16) & 1u);   // round-to-nearest-even
  return (unsigned short)(r >> 16);
}

__device__ __forceinline__ v8f wmma_bf16(v8u a, v8u b, v8f c) {
  return __builtin_amdgcn_wmma_f32_16x16x32_bf16(
      false, __builtin_bit_cast(v16bf, a),
      false, __builtin_bit_cast(v16bf, b),
      (short)0, c, false, false);
}

// A operand (16x32 bf16, row-major source, lda elems). lane: row=lane&15,
// hi=lane>>4 selects K-subblock (contiguous, matching B loaders):
//   VGPR p(0..7): K = k0 + hi*16 + 2p
__device__ __forceinline__ v8u load_a(const unsigned short* A, int lda,
                                      int row, int k0, int hi) {
  const unsigned short* p = A + (size_t)row * lda + k0 + hi * 16;
  v8u a;
#pragma unroll
  for (int j = 0; j < 8; ++j) a[j] = *(const unsigned int*)(p + 2 * j);
  return a;
}

// B operand from an (Nout,K) row-major weight ("NT"): B[k,n] = W[n,k].
// lane: n=col, K = k0 + hi*16 + j  (pairs contiguous in the W row).
__device__ __forceinline__ v8u load_b_nt(const unsigned short* W, int ldb,
                                         int col, int k0, int hi) {
  const unsigned short* p = W + (size_t)col * ldb + k0 + hi * 16;
  v8u b;
#pragma unroll
  for (int j = 0; j < 8; ++j) b[j] = *(const unsigned int*)(p + 2 * j);
  return b;
}

// B operand from a (K,N) row-major matrix ("NN"): strided per-row gathers.
__device__ __forceinline__ v8u load_b_nn(const unsigned short* Z, int ldz,
                                         int col, int k0, int hi) {
  v8u b;
#pragma unroll
  for (int j = 0; j < 8; ++j) {
    int k = k0 + hi * 16 + 2 * j;
    unsigned lo = Z[(size_t)k * ldz + col];
    unsigned h2 = Z[(size_t)(k + 1) * ldz + col];
    b[j] = lo | (h2 << 16);
  }
  return b;
}

// ----- small elementwise kernels ------------------------------------------

__global__ void cvt_bf16_kernel(const float* s, unsigned short* d, int n) {
  int i = blockIdx.x * 256 + threadIdx.x;
  if (i < n) d[i] = f2bf(s[i]);
}

__global__ void adjpad_kernel(const float* adj, unsigned short* dst) {
  int i = blockIdx.x * 256 + threadIdx.x;
  if (i >= NP * NP) return;
  int w = i / NP, v = i % NP;
  float val = (w < Nn && v < Nn) ? adj[w * Nn + v] : 0.f;
  dst[i] = f2bf(val);
}

// s[n,b,:] = h[b,:,n] + embed[n,:]*mask_last_tp[b,0,n]   row = n*B+b
__global__ __launch_bounds__(256) void builds_kernel(
    const float* h, const float* embed, const float* mlt,
    float* s32, unsigned short* sb) {
  int row = blockIdx.x;
  int n = row >> 8, b = row & 255, u = threadIdx.x;
  float val = h[((size_t)b * NUc + u) * Nn + n] +
              embed[(size_t)n * NUc + u] * mlt[(size_t)b * Nn + n];
  size_t idx = (size_t)row * NUc + u;
  s32[idx] = val;
  sb[idx]  = f2bf(val);
}

// x -> z block0 (c<64) + zero pad cols for c in [0,320)
__global__ void packx_kernel(const float* x, unsigned short* z) {
  int v = threadIdx.x;            // 0..223
  int c = blockIdx.x;             // 0..319
  int b = blockIdx.y;
  size_t zi = (size_t)b * KCc * NP + (size_t)c * NP + v;
  if (c < DIN) {
    float val = (v < Nn) ? x[((size_t)b * DIN + c) * Nn + v] : 0.f;
    z[zi] = f2bf(val);
  } else if (v >= Nn) {
    z[zi] = 0;                    // pad cols of h-part (values come from ln2)
  }
}

// ----- generic NT GEMM (K=256):  C = epi(A(M,256) @ W(Nout,256)^T + bias) --
// A tile cached in 64 VGPRs, reused across 4 n-tiles per wave.
// mode 0: store bf16          (QKV)
// mode 1: +res, store fp32    (out-proj / FFN2 pre-LN)
// mode 2: relu, store bf16    (FFN1)
__global__ __launch_bounds__(32) void gemm_nt_kernel(
    const unsigned short* A, const unsigned short* W,
    const float* bias, const float* res, float* outf, unsigned short* outb,
    int Nout, int mode) {
  int m0   = blockIdx.x * 16;
  int ng   = blockIdx.y * 4;          // group of 4 n-tiles
  int lane = threadIdx.x & 31, hi = lane >> 4;
  int arow = m0 + (lane & 15);
  v8u areg[8];
#pragma unroll
  for (int s = 0; s < 8; ++s) areg[s] = load_a(A, 256, arow, s * 32, hi);
#pragma unroll
  for (int t = 0; t < 4; ++t) {
    int n0  = (ng + t) * 16;
    int col = n0 + (lane & 15);
    v8f acc = {0.f, 0.f, 0.f, 0.f, 0.f, 0.f, 0.f, 0.f};
#pragma unroll
    for (int s = 0; s < 8; ++s)
      acc = wmma_bf16(areg[s], load_b_nt(W, 256, col, s * 32, hi), acc);
    float bv = bias[col];
#pragma unroll
    for (int r = 0; r < 8; ++r) {
      int m = m0 + r + hi * 8;
      size_t idx = (size_t)m * Nout + col;
      float v = acc[r] + bv;
      if (mode == 0)      outb[idx] = f2bf(v);
      else if (mode == 1) outf[idx] = v + res[idx];
      else                outb[idx] = f2bf(fmaxf(v, 0.f));
    }
  }
}

// ----- fused attention: scores(WMMA) -> softmax(LDS) -> *V(WMMA) ----------
__global__ __launch_bounds__(256) void attn_kernel(const unsigned short* qkv,
                                                   unsigned short* ob) {
  __shared__ unsigned short Qs[16 * 256];
  __shared__ float          Ss[16 * NP];
  __shared__ unsigned short Ps[16 * NP];
  int b = blockIdx.y, n0 = blockIdx.x * 16;
  int tid = threadIdx.x, wave = tid >> 5, lane = tid & 31, hi = lane >> 4;

  for (int i = tid; i < 16 * 256; i += 256) {           // load Q tile (bf16)
    int rr = i >> 8, cc = i & 255, n = n0 + rr;
    Qs[i] = (n < Nn) ? qkv[((size_t)n * Bb + b) * 768 + cc] : (unsigned short)0;
  }
  __syncthreads();

  for (int t = wave; t < 13; t += 8) {                  // scores S = QK^T/16
    int m0 = t * 16, mcol = m0 + (lane & 15);
    v8f acc = {0.f, 0.f, 0.f, 0.f, 0.f, 0.f, 0.f, 0.f};
    for (int k0 = 0; k0 < 256; k0 += 32) {
      v8u a = load_a(Qs, 256, lane & 15, k0, hi);
      v8u bb = {0, 0, 0, 0, 0, 0, 0, 0};
      if (mcol < Nn) {
        const unsigned short* p =
            qkv + ((size_t)mcol * Bb + b) * 768 + 256 + k0 + hi * 16;
#pragma unroll
        for (int j = 0; j < 8; ++j) bb[j] = *(const unsigned int*)(p + 2 * j);
      }
      acc = wmma_bf16(a, bb, acc);
    }
#pragma unroll
    for (int r = 0; r < 8; ++r)
      Ss[(r + hi * 8) * NP + mcol] = acc[r] * 0.0625f;
  }
  __syncthreads();
  for (int i = tid; i < 16 * 17; i += 256) {            // mask cols 207..223
    int rr = i / 17, cc = Nn + (i % 17);
    Ss[rr * NP + cc] = -1e30f;
  }
  __syncthreads();
  if (tid < 16) {                                       // softmax per row
    float mx = -1e30f;
    for (int c = 0; c < NP; ++c) mx = fmaxf(mx, Ss[tid * NP + c]);
    float sum = 0.f;
    for (int c = 0; c < NP; ++c) {
      float e = __expf(Ss[tid * NP + c] - mx);
      Ss[tid * NP + c] = e;
      sum += e;
    }
    float inv = 1.f / sum;
    for (int c = 0; c < NP; ++c) Ps[tid * NP + c] = f2bf(Ss[tid * NP + c] * inv);
  }
  __syncthreads();

  for (int t2 = wave * 2; t2 < wave * 2 + 2; ++t2) {    // O = P @ V
    int d0 = t2 * 16, col = d0 + (lane & 15);
    v8f acc = {0.f, 0.f, 0.f, 0.f, 0.f, 0.f, 0.f, 0.f};
    for (int k0 = 0; k0 < NP; k0 += 32) {
      v8u a = load_a(Ps, NP, lane & 15, k0, hi);
      v8u bb;
#pragma unroll
      for (int j = 0; j < 8; ++j) {
        int m = k0 + hi * 16 + 2 * j;
        unsigned lo = (m < Nn)     ? qkv[((size_t)m * Bb + b) * 768 + 512 + col] : 0u;
        unsigned h2 = (m + 1 < Nn) ? qkv[((size_t)(m + 1) * Bb + b) * 768 + 512 + col] : 0u;
        bb[j] = lo | (h2 << 16);
      }
      acc = wmma_bf16(a, bb, acc);
    }
#pragma unroll
    for (int r = 0; r < 8; ++r) {
      int n = n0 + r + hi * 8;
      if (n < Nn) ob[((size_t)n * Bb + b) * 256 + col] = f2bf(acc[r]);
    }
  }
}

// ----- LayerNorms ----------------------------------------------------------
__global__ __launch_bounds__(256) void ln1_kernel(const float* X, const float* w,
                                                  const float* bi, float* of,
                                                  unsigned short* obf) {
  __shared__ float sm[256];
  int row = blockIdx.x, t = threadIdx.x;
  float x = X[(size_t)row * 256 + t];
  sm[t] = x; __syncthreads();
  for (int s = 128; s > 0; s >>= 1) { if (t < s) sm[t] += sm[t + s]; __syncthreads(); }
  float mu = sm[0] * (1.f / 256.f); __syncthreads();
  float d = x - mu;
  sm[t] = d * d; __syncthreads();
  for (int s = 128; s > 0; s >>= 1) { if (t < s) sm[t] += sm[t + s]; __syncthreads(); }
  float var = sm[0] * (1.f / 256.f);
  float y = d * rsqrtf(var + 1e-5f) * w[t] + bi[t];
  of[(size_t)row * 256 + t]  = y;
  obf[(size_t)row * 256 + t] = f2bf(y);
}

// LN2 + tanh + h-update, writes h_new (f32, (B,NU,N)) and the h-part of z.
__global__ __launch_bounds__(256) void ln2_kernel(const float* X, const float* w,
                                                  const float* bi, const float* hold,
                                                  const float* dt, float* hn,
                                                  unsigned short* z) {
  __shared__ float sm[256];
  int row = blockIdx.x, t = threadIdx.x;
  float x = X[(size_t)row * 256 + t];
  sm[t] = x; __syncthreads();
  for (int s = 128; s > 0; s >>= 1) { if (t < s) sm[t] += sm[t + s]; __syncthreads(); }
  float mu = sm[0] * (1.f / 256.f); __syncthreads();
  float d = x - mu;
  sm[t] = d * d; __syncthreads();
  for (int s = 128; s > 0; s >>= 1) { if (t < s) sm[t] += sm[t + s]; __syncthreads(); }
  float var = sm[0] * (1.f / 256.f);
  float y = d * rsqrtf(var + 1e-5f) * w[t] + bi[t];
  float e = tanhf(y);
  int n = row >> 8, b = row & 255;
  float dv = dt[b];
  float hv = sqrtf(1.f - dv) * hold[((size_t)b * NUc + t) * Nn + n] - sqrtf(dv) * e;
  hn[((size_t)b * NUc + t) * Nn + n] = hv;
  z[(size_t)b * KCc * NP + (size_t)(DIN + t) * NP + n] = f2bf(hv);
}

// ----- diffusion round: z[:, dst:dst+320, :] = z[:, src:src+320, :] @ adj^T
// A tile (16x224) cached in 56 VGPRs, reused for all 14 w-tiles.
__global__ __launch_bounds__(32) void diffusion_kernel(unsigned short* z,
                                                       const unsigned short* adjp,
                                                       int src_c, int dst_c) {
  int ct = blockIdx.x;   // 0..19  (16-row channel tile)
  int b  = blockIdx.y;
  int lane = threadIdx.x, hi = lane >> 4;
  const unsigned short* A = z + (size_t)b * KCc * NP + (size_t)(src_c + ct * 16) * NP;
  v8u areg[7];
#pragma unroll
  for (int s = 0; s < 7; ++s) areg[s] = load_a(A, NP, lane & 15, s * 32, hi);
  unsigned short* C = z + (size_t)b * KCc * NP + (size_t)dst_c * NP;
  for (int wt = 0; wt < 14; ++wt) {    // padded rows of adjp are zero
    int col = wt * 16 + (lane & 15);
    v8f acc = {0.f, 0.f, 0.f, 0.f, 0.f, 0.f, 0.f, 0.f};
#pragma unroll
    for (int s = 0; s < 7; ++s)
      acc = wmma_bf16(areg[s], load_b_nt(adjp, NP, col, s * 32, hi), acc);
#pragma unroll
    for (int r = 0; r < 8; ++r) {
      int row = ct * 16 + r + hi * 8;
      C[(size_t)row * NP + col] = f2bf(acc[r]);
    }
  }
}

// ----- fused gates: u & c over a quad of m-tiles; one z load -> 8 WMMAs ----
__global__ __launch_bounds__(32) void gate_kernel(const unsigned short* z,
                                                  const unsigned short* wu,
                                                  const unsigned short* wc,
                                                  const float* bu, const float* bc,
                                                  const float* hn, float* out) {
  int nt = blockIdx.x;   // 0..12
  int mq = blockIdx.y;   // 0..3 (quad of 16-row m-tiles)
  int b  = blockIdx.z;
  int lane = threadIdx.x, hi = lane >> 4;
  const unsigned short* Z = z + (size_t)b * KCc * NP;
  int col = nt * 16 + (lane & 15);
  v8f au[4], ac[4];
#pragma unroll
  for (int q = 0; q < 4; ++q) {
    au[q] = (v8f){0.f, 0.f, 0.f, 0.f, 0.f, 0.f, 0.f, 0.f};
    ac[q] = (v8f){0.f, 0.f, 0.f, 0.f, 0.f, 0.f, 0.f, 0.f};
  }
  for (int k0 = 0; k0 < KCc; k0 += 32) {
    v8u bz = load_b_nn(Z, NP, col, k0, hi);
#pragma unroll
    for (int q = 0; q < 4; ++q) {
      int wrow = (mq * 4 + q) * 16 + (lane & 15);
      v8u a1 = load_a(wu, KCc, wrow, k0, hi);
      v8u a2 = load_a(wc, KCc, wrow, k0, hi);
      au[q] = wmma_bf16(a1, bz, au[q]);
      ac[q] = wmma_bf16(a2, bz, ac[q]);
    }
  }
  if (col < Nn) {
#pragma unroll
    for (int q = 0; q < 4; ++q) {
#pragma unroll
      for (int r = 0; r < 8; ++r) {
        int o = (mq * 4 + q) * 16 + r + hi * 8;
        float uv = 1.f / (1.f + __expf(-(au[q][r] + bu[o])));
        float cv = tanhf(ac[q][r] + bc[o]);
        float hv = hn[((size_t)b * NUc + o) * Nn + col];
        out[((size_t)b * NUc + o) * Nn + col] = uv * hv + (1.f - uv) * cv;
      }
    }
  }
}

// ---------------------------------------------------------------------------

extern "C" void kernel_launch(void* const* d_in, const int* in_sizes, int n_in,
                              void* d_out, int out_size, void* d_ws, size_t ws_size,
                              hipStream_t stream) {
  (void)in_sizes; (void)n_in; (void)out_size; (void)ws_size;
  const float* x     = (const float*)d_in[0];
  const float* dt    = (const float*)d_in[1];
  const float* h     = (const float*)d_in[2];
  const float* adj   = (const float*)d_in[3];
  const float* mlt   = (const float*)d_in[5];
  const float* embed = (const float*)d_in[8];
  const float* ipw = (const float*)d_in[9];  const float* ipb = (const float*)d_in[10];
  const float* opw = (const float*)d_in[11]; const float* opb = (const float*)d_in[12];
  const float* l1w = (const float*)d_in[13]; const float* l1b = (const float*)d_in[14];
  const float* l2w = (const float*)d_in[15]; const float* l2b = (const float*)d_in[16];
  const float* f1w = (const float*)d_in[17]; const float* f1b = (const float*)d_in[18];
  const float* f2w = (const float*)d_in[19]; const float* f2b = (const float*)d_in[20];
  const float* Wu  = (const float*)d_in[21]; const float* bu  = (const float*)d_in[22];
  const float* Wc  = (const float*)d_in[23]; const float* bc  = (const float*)d_in[24];
  float* out = (float*)d_out;

  char* ws = (char*)d_ws;
  size_t off = 0;
  auto alloc = [&](size_t bytes) { size_t o = off; off = (off + bytes + 255) & ~(size_t)255; return o; };
  unsigned short* sb   = (unsigned short*)(ws + alloc((size_t)NBr * 256 * 2));  // also fb
  float*          s32  = (float*)(ws + alloc((size_t)NBr * 256 * 4));           // also s1f
  unsigned short* qkvb = (unsigned short*)(ws + alloc((size_t)NBr * 768 * 2));  // also hn
  unsigned short* ob   = (unsigned short*)(ws + alloc((size_t)NBr * 256 * 2));
  float*          r1   = (float*)(ws + alloc((size_t)NBr * 256 * 4));           // also r2
  unsigned short* s1b  = (unsigned short*)(ws + alloc((size_t)NBr * 256 * 2));
  unsigned short* zb   = (unsigned short*)(ws + alloc((size_t)Bb * KCc * NP * 2));
  unsigned short* adjp = (unsigned short*)(ws + alloc((size_t)NP * NP * 2));
  unsigned short* ipwb = (unsigned short*)(ws + alloc((size_t)768 * 256 * 2));
  unsigned short* opwb = (unsigned short*)(ws + alloc((size_t)256 * 256 * 2));
  unsigned short* f1wb = (unsigned short*)(ws + alloc((size_t)256 * 256 * 2));
  unsigned short* f2wb = (unsigned short*)(ws + alloc((size_t)256 * 256 * 2));
  unsigned short* wub  = (unsigned short*)(ws + alloc((size_t)256 * KCc * 2));
  unsigned short* wcb  = (unsigned short*)(ws + alloc((size_t)256 * KCc * 2));
  float*          s1f  = s32;           // lifetimes disjoint (aliased regions)
  float*          r2   = r1;
  float*          hn   = (float*)qkvb;
  unsigned short* fb   = sb;

  auto cvt = [&](const float* s, unsigned short* d, int n) {
    cvt_bf16_kernel<<<(n + 255) / 256, 256, 0, stream>>>(s, d, n);
  };
  cvt(ipw, ipwb, 768 * 256);
  cvt(opw, opwb, 256 * 256);
  cvt(f1w, f1wb, 256 * 256);
  cvt(f2w, f2wb, 256 * 256);
  cvt(Wu,  wub,  256 * KCc);
  cvt(Wc,  wcb,  256 * KCc);
  adjpad_kernel<<<(NP * NP + 255) / 256, 256, 0, stream>>>(adj, adjp);

  builds_kernel<<<NBr, 256, 0, stream>>>(h, embed, mlt, s32, sb);
  // QKV: (52992,256) @ (768,256)^T -> bf16
  gemm_nt_kernel<<<dim3(NBr / 16, 12), 32, 0, stream>>>(
      sb, ipwb, ipb, nullptr, nullptr, qkvb, 768, 0);
  attn_kernel<<<dim3(13, Bb), 256, 0, stream>>>(qkvb, ob);
  // out-proj + residual -> r1 (fp32)
  gemm_nt_kernel<<<dim3(NBr / 16, 4), 32, 0, stream>>>(
      ob, opwb, opb, s32, r1, nullptr, 256, 1);
  ln1_kernel<<<NBr, 256, 0, stream>>>(r1, l1w, l1b, s1f, s1b);
  // FFN1 (relu, bf16)
  gemm_nt_kernel<<<dim3(NBr / 16, 4), 32, 0, stream>>>(
      s1b, f1wb, f1b, nullptr, nullptr, fb, 256, 2);
  // FFN2 + residual -> r2 (fp32)
  gemm_nt_kernel<<<dim3(NBr / 16, 4), 32, 0, stream>>>(
      fb, f2wb, f2b, s1f, r2, nullptr, 256, 1);
  // LN2 + tanh + h update (writes hn and z h-part)
  ln2_kernel<<<NBr, 256, 0, stream>>>(r2, l2w, l2b, h, dt, hn, zb);
  // x -> z block0 + pads
  packx_kernel<<<dim3(320, Bb), NP, 0, stream>>>(x, zb);
  // order-3 diffusion, in place in z (A cached in registers across w-tiles)
  diffusion_kernel<<<dim3(20, Bb), 32, 0, stream>>>(zb, adjp, 0,   320);
  diffusion_kernel<<<dim3(20, Bb), 32, 0, stream>>>(zb, adjp, 320, 640);
  diffusion_kernel<<<dim3(20, Bb), 32, 0, stream>>>(zb, adjp, 640, 960);
  // fused u/c gates + GRU combine -> d_out
  gate_kernel<<<dim3(13, 4, Bb), 32, 0, stream>>>(zb, wub, wcb, bu, bc, hn, out);
}
